// Bottleneck_49039936586369
// MI455X (gfx1250) — matrix operand order repaired and checked by reading, not compiled, not run on hardware
//
#include <hip/hip_runtime.h>
#include <math.h>

// ---------------------------------------------------------------------------
// Problem constants (reference: B=8, C=256, H=W=32, 8 heads of 32ch, 4 groups of 64ch)
// ---------------------------------------------------------------------------
static constexpr int BATCH = 8;
static constexpr int C     = 256;
static constexpr int HH    = 32;
static constexpr int WW    = 32;
static constexpr int HW    = HH * WW;     // 1024
static constexpr int NH    = 8;
static constexpr int HC    = C / NH;      // 32
static constexpr int NG    = 4;
static constexpr int GC    = C / NG;      // 64

typedef __attribute__((ext_vector_type(16))) _Float16 v16h;
typedef __attribute__((ext_vector_type(8)))  _Float16 v8h;
typedef __attribute__((ext_vector_type(8)))  float    v8f;
typedef int v4i __attribute__((vector_size(4 * sizeof(int))));

// Async-to-LDS path (gfx1250); guarded since the builtin is toolchain-dependent.
#if defined(__gfx1250__) && __has_builtin(__builtin_amdgcn_global_load_async_to_lds_b128) && \
    __has_builtin(__builtin_amdgcn_s_wait_asynccnt)
#define HAVE_ASYNC_LDS 1
#else
#define HAVE_ASYNC_LDS 0
#endif

// ---------------------------------------------------------------------------
// WMMA operand loaders (CDNA5 16x16x32 f16 layouts, ISA 7.12.2)
//  A (16x32, MxK): lane<16 -> M=lane,  elems 0..7 = K 0..7,  8..15 = K 16..23
//                  lane>=16 -> M=lane-16, elems 0..7 = K 8..15, 8..15 = K 24..31
//  B (32x16, KxN): lane<16 -> N=lane,  elems 0..7 = K 0..7,  8..15 = K 16..23
//                  lane>=16 -> N=lane-16, elems 0..7 = K 8..15, 8..15 = K 24..31
//  C/D (16x16):    vgpr i, lane<16: (M=i, N=lane); lane>=16: (M=i+8, N=lane-16)
// ---------------------------------------------------------------------------
__device__ __forceinline__ v16h pack16(const v8h lo, const v8h hi) {
  v16h r;
#pragma unroll
  for (int i = 0; i < 8; ++i) { r[i] = lo[i]; r[i + 8] = hi[i]; }
  return r;
}

__device__ __forceinline__ v16h load_a16x32(const _Float16* p) {
  const int hf = (threadIdx.x >> 4) & 1;
  const v8h lo = *(const v8h*)(p + hf * 8);
  const v8h hi = *(const v8h*)(p + 16 + hf * 8);
  return pack16(lo, hi);
}

__device__ __forceinline__ v16h load_b32x16(const _Float16* p) {
  const int hf = (threadIdx.x >> 4) & 1;
  const v8h lo = *(const v8h*)(p + hf * 8);
  const v8h hi = *(const v8h*)(p + 16 + hf * 8);
  return pack16(lo, hi);
}

#define WMMA_F16(a, b, c) \
  __builtin_amdgcn_wmma_f32_16x16x32_f16(false, (a), false, (b), (short)0, (c), false, false)

// ---------------------------------------------------------------------------
// Conversion kernels
// ---------------------------------------------------------------------------
__global__ void __launch_bounds__(256) cvt_f16_kernel(const float* __restrict__ src,
                                                      _Float16* __restrict__ dst, int n) {
  int i = blockIdx.x * blockDim.x + threadIdx.x;
  if (i < n) dst[i] = (_Float16)src[i];
}

// x (B,C,HW) f32  ->  xT16 (B,HW,C) f16   (token-major for WMMA A operands)
__global__ void __launch_bounds__(256) cvt_xT_kernel(const float* __restrict__ x,
                                                     _Float16* __restrict__ xT16) {
  int i = blockIdx.x * blockDim.x + threadIdx.x;
  if (i >= BATCH * C * HW) return;
  const int b  = i / (C * HW);
  const int r  = i - b * (C * HW);
  const int c  = r / HW;
  const int hw = r - c * HW;
  xT16[((size_t)b * HW + hw) * C + c] = (_Float16)x[i];
}

// ---------------------------------------------------------------------------
// Projection GEMMs:  out[n,o] = sum_c A_T16[b,n,c] * W16[o,c] + bias[o]
// One wave per 16x64 output tile: A operand reused across 4 column tiles,
// 32 WMMAs per wave, K = 256.
// ---------------------------------------------------------------------------
__device__ __forceinline__ void gemm_16x64(const _Float16* __restrict__ arow,
                                           const _Float16* __restrict__ brow,
                                           v8f acc[4]) {
#pragma unroll
  for (int k0 = 0; k0 < C; k0 += 32) {
    const v16h a = load_a16x32(arow + k0);
#pragma unroll
    for (int t = 0; t < 4; ++t)
      acc[t] = WMMA_F16(a, load_b32x16(brow + (size_t)(t * 16) * C + k0), acc[t]);
  }
}

__global__ void __launch_bounds__(32) qproj_kernel(const _Float16* __restrict__ xT16,
                                                   const _Float16* __restrict__ W16,
                                                   const float* __restrict__ bias,
                                                   float* __restrict__ qT,
                                                   _Float16* __restrict__ qT16) {
  const int lane = threadIdx.x, r = lane & 15, hf = lane >> 4;
  const int n0 = blockIdx.x * 16, o0 = blockIdx.y * 64, b = blockIdx.z;
  const _Float16* arow = xT16 + ((size_t)b * HW + n0 + r) * C;
  const _Float16* brow = W16 + (size_t)(o0 + r) * C;
  v8f acc[4] = {};
  gemm_16x64(arow, brow, acc);
#pragma unroll
  for (int t = 0; t < 4; ++t) {
    const int col = o0 + t * 16 + r;
    const float bb = bias[col];
#pragma unroll
    for (int i = 0; i < 8; ++i) {
      const int row = n0 + i + 8 * hf;
      const float v = acc[t][i] + bb;
      qT[((size_t)b * HW + row) * C + col]   = v;
      qT16[((size_t)b * HW + row) * C + col] = (_Float16)v;
    }
  }
}

__global__ void __launch_bounds__(32) kproj_kernel(const _Float16* __restrict__ xsT16,
                                                   const _Float16* __restrict__ W16,
                                                   const float* __restrict__ bias,
                                                   _Float16* __restrict__ kT16) {
  const int lane = threadIdx.x, r = lane & 15, hf = lane >> 4;
  const int n0 = blockIdx.x * 16, o0 = blockIdx.y * 64, b = blockIdx.z;
  const _Float16* arow = xsT16 + ((size_t)b * HW + n0 + r) * C;
  const _Float16* brow = W16 + (size_t)(o0 + r) * C;
  v8f acc[4] = {};
  gemm_16x64(arow, brow, acc);
#pragma unroll
  for (int t = 0; t < 4; ++t) {
    const int col = o0 + t * 16 + r;
    const float bb = bias[col];
#pragma unroll
    for (int i = 0; i < 8; ++i) {
      const int row = n0 + i + 8 * hf;
      kT16[((size_t)b * HW + row) * C + col] = (_Float16)(acc[t][i] + bb);
    }
  }
}

// v stored channel-major (B, C, n) so the PV WMMA's B operand is contiguous
__global__ void __launch_bounds__(32) vproj_kernel(const _Float16* __restrict__ xsT16,
                                                   const _Float16* __restrict__ W16,
                                                   const float* __restrict__ bias,
                                                   _Float16* __restrict__ vcm16) {
  const int lane = threadIdx.x, r = lane & 15, hf = lane >> 4;
  const int n0 = blockIdx.x * 16, o0 = blockIdx.y * 64, b = blockIdx.z;
  const _Float16* arow = xsT16 + ((size_t)b * HW + n0 + r) * C;
  const _Float16* brow = W16 + (size_t)(o0 + r) * C;
  v8f acc[4] = {};
  gemm_16x64(arow, brow, acc);
#pragma unroll
  for (int t = 0; t < 4; ++t) {
    const int col = o0 + t * 16 + r;
    const float bb = bias[col];
#pragma unroll
    for (int i = 0; i < 8; ++i) {
      const int row = n0 + i + 8 * hf;
      vcm16[((size_t)b * C + col) * HW + row] = (_Float16)(acc[t][i] + bb);
    }
  }
}

__global__ void __launch_bounds__(32) oproj_kernel(const _Float16* __restrict__ outT16,
                                                   const _Float16* __restrict__ W16,
                                                   const float* __restrict__ bias,
                                                   float* __restrict__ y) {
  const int lane = threadIdx.x, r = lane & 15, hf = lane >> 4;
  const int n0 = blockIdx.x * 16, o0 = blockIdx.y * 64, b = blockIdx.z;
  const _Float16* arow = outT16 + ((size_t)b * HW + n0 + r) * C;
  const _Float16* brow = W16 + (size_t)(o0 + r) * C;
  v8f acc[4] = {};
  gemm_16x64(arow, brow, acc);
#pragma unroll
  for (int t = 0; t < 4; ++t) {
    const int col = o0 + t * 16 + r;
    const float bb = bias[col];
#pragma unroll
    for (int i = 0; i < 8; ++i) {
      const int row = n0 + i + 8 * hf;  // hw
      y[((size_t)b * C + col) * HW + row] = acc[t][i] + bb;
    }
  }
}

// ---------------------------------------------------------------------------
// Offset head: dwconv3x3 -> LN(channels) -> GELU(exact) -> pw(2xGC) -> tanh*4/31 + ref grid
// ---------------------------------------------------------------------------
__global__ void __launch_bounds__(256) offset_kernel(const float* __restrict__ qT,
                                                     const float* __restrict__ dw_w,
                                                     const float* __restrict__ dw_b,
                                                     const float* __restrict__ ln_g,
                                                     const float* __restrict__ ln_b,
                                                     const float* __restrict__ pw_w,
                                                     float* __restrict__ pos) {
  int idx = blockIdx.x * blockDim.x + threadIdx.x;
  if (idx >= BATCH * NG * HW) return;
  const int bg = idx / HW, hw = idx - bg * HW;
  const int b = bg >> 2, g = bg & 3;
  const int y = hw >> 5, x = hw & 31;

  float o[GC];
  float mu = 0.f;
#pragma unroll 2
  for (int c = 0; c < GC; ++c) {
    float acc = dw_b[c];
#pragma unroll
    for (int dy = -1; dy <= 1; ++dy)
#pragma unroll
      for (int dx = -1; dx <= 1; ++dx) {
        const int yy = y + dy, xx = x + dx;
        if (yy >= 0 && yy < HH && xx >= 0 && xx < WW)
          acc += dw_w[c * 9 + (dy + 1) * 3 + (dx + 1)] *
                 qT[((size_t)b * HW + (yy * WW + xx)) * C + g * GC + c];
      }
    o[c] = acc;
    mu += acc;
  }
  mu *= (1.0f / GC);
  float var = 0.f;
#pragma unroll 4
  for (int c = 0; c < GC; ++c) { const float d = o[c] - mu; var += d * d; }
  var *= (1.0f / GC);
  const float rstd = rsqrtf(var + 1e-5f);
  float offy = 0.f, offx = 0.f;
#pragma unroll 4
  for (int c = 0; c < GC; ++c) {
    float v = (o[c] - mu) * rstd * ln_g[c] + ln_b[c];
    v = 0.5f * v * (1.0f + erff(v * 0.70710678118654752f));  // exact GELU
    offy += pw_w[c] * v;
    offx += pw_w[GC + c] * v;
  }
  const float rngf = 4.0f / 31.0f;  // OFFSET_RANGE_FACTOR / (H-1)
  offy = tanhf(offy) * rngf;
  offx = tanhf(offx) * rngf;
  const float ry = ((y + 0.5f) / 31.0f) * 2.0f - 1.0f;
  const float rx = ((x + 0.5f) / 31.0f) * 2.0f - 1.0f;
  pos[(size_t)idx * 2 + 0] = offy + ry;
  pos[(size_t)idx * 2 + 1] = offx + rx;
}

// ---------------------------------------------------------------------------
// Bilinear grid sample (align_corners=True, zeros padding) -> token-major f16
// ---------------------------------------------------------------------------
__global__ void __launch_bounds__(256) sample_kernel(const float* __restrict__ x,
                                                     const float* __restrict__ pos,
                                                     _Float16* __restrict__ xsT16) {
  int idx = blockIdx.x * blockDim.x + threadIdx.x;
  if (idx >= BATCH * NG * HW) return;
  const int bg = idx / HW, n = idx - bg * HW;
  const int b = bg >> 2, g = bg & 3;
  const float py = pos[(size_t)idx * 2 + 0];
  const float px = pos[(size_t)idx * 2 + 1];
  const float fy = (py + 1.0f) * 0.5f * (HH - 1);
  const float fx = (px + 1.0f) * 0.5f * (WW - 1);
  const float y0f = floorf(fy), x0f = floorf(fx);
  const int y0 = (int)y0f, x0 = (int)x0f;
  const float yw = fy - y0f, xw = fx - x0f;
  const int y1 = y0 + 1, x1 = x0 + 1;
  float w00 = (1 - yw) * (1 - xw), w01 = (1 - yw) * xw;
  float w10 = yw * (1 - xw), w11 = yw * xw;
  w00 *= (y0 >= 0 && y0 < HH && x0 >= 0 && x0 < WW) ? 1.f : 0.f;
  w01 *= (y0 >= 0 && y0 < HH && x1 >= 0 && x1 < WW) ? 1.f : 0.f;
  w10 *= (y1 >= 0 && y1 < HH && x0 >= 0 && x0 < WW) ? 1.f : 0.f;
  w11 *= (y1 >= 0 && y1 < HH && x1 >= 0 && x1 < WW) ? 1.f : 0.f;
  const int iy0 = min(max(y0, 0), HH - 1), iy1 = min(max(y1, 0), HH - 1);
  const int ix0 = min(max(x0, 0), WW - 1), ix1 = min(max(x1, 0), WW - 1);
  const int i00 = iy0 * WW + ix0, i01 = iy0 * WW + ix1;
  const int i10 = iy1 * WW + ix0, i11 = iy1 * WW + ix1;
  const float* base = x + ((size_t)b * C + g * GC) * HW;
  _Float16* out = xsT16 + ((size_t)b * HW + n) * C + g * GC;
#pragma unroll 4
  for (int c = 0; c < GC; ++c) {
    const float* pc = base + (size_t)c * HW;
    out[c] = (_Float16)(w00 * pc[i00] + w01 * pc[i01] + w10 * pc[i10] + w11 * pc[i11]);
  }
}

// ---------------------------------------------------------------------------
// Fused flash attention + lepe (depthwise 3x3 on q) residual.
// One wave per (batch, head, 16-row q-tile). Scores never leave registers; P is
// transposed D->A layout through LDS. K/V chunks (32 keys x 64B) are prefetched
// into LDS with double-buffered global_load_async_to_lds_b128 when available.
// ---------------------------------------------------------------------------
__device__ __forceinline__ float lepe_val(const float* __restrict__ qT,
                                          const float* __restrict__ rw,
                                          const float* __restrict__ rb,
                                          int b, int ch, int m) {
  const int y = m >> 5, x = m & 31;
  float acc = rb[ch];
#pragma unroll
  for (int dy = -1; dy <= 1; ++dy)
#pragma unroll
    for (int dx = -1; dx <= 1; ++dx) {
      const int yy = y + dy, xx = x + dx;
      if (yy >= 0 && yy < HH && xx >= 0 && xx < WW)
        acc += rw[ch * 9 + (dy + 1) * 3 + (dx + 1)] *
               qT[((size_t)b * HW + (yy * WW + xx)) * C + ch];
    }
  return acc;
}

#if HAVE_ASYNC_LDS
// Copy 32 rows x 64 bytes (row_stride in halfs) into LDS: 4 async b128 issues.
// Builtin signature (from hipcc diagnostic): arg0 = v4i in AS1 (global src).
__device__ __forceinline__ void stage_chunk(const _Float16* src, size_t row_stride,
                                            _Float16* dst, int lane) {
  const int part = lane & 3;   // 16B granule within a 64B row
  const int rr   = lane >> 2;  // 8 rows per issue
#pragma unroll
  for (int j = 0; j < 4; ++j) {
    const int row = j * 8 + rr;
    __builtin_amdgcn_global_load_async_to_lds_b128(
        (__attribute__((address_space(1))) v4i*)(src + (size_t)row * row_stride + part * 8),
        (__attribute__((address_space(3))) v4i*)(dst + row * 32 + part * 8),
        0, 0);
  }
}
#endif

__global__ void __launch_bounds__(32) attn_kernel(const _Float16* __restrict__ qT16,
                                                  const _Float16* __restrict__ kT16,
                                                  const _Float16* __restrict__ vcm16,
                                                  const float* __restrict__ qT,
                                                  const float* __restrict__ rpe_w,
                                                  const float* __restrict__ rpe_b,
                                                  _Float16* __restrict__ outT16) {
  const int lane = threadIdx.x, r = lane & 15, hf = lane >> 4;
  const int m0 = blockIdx.x * 16, h = blockIdx.y, b = blockIdx.z;
  const float scale = 0.17677669529663687f;  // HC^-0.5

  __shared__ __align__(16) _Float16 ldsP[16 * 32];
#if HAVE_ASYNC_LDS
  __shared__ __align__(16) _Float16 ldsK[2][32 * 32];
  __shared__ __align__(16) _Float16 ldsV[2][32 * 32];
#endif

  const _Float16* kBase = kT16 + (size_t)b * HW * C + h * HC;       // + n*C per key
  const _Float16* vBase = vcm16 + ((size_t)b * C + h * HC) * HW;    // + c*HW, + n

  // q tile: rows m0..m0+15, K = 32 head channels (single A operand, lives in regs)
  const v16h qa = load_a16x32(qT16 + ((size_t)b * HW + m0 + r) * C + h * HC);

  v8f acc0 = {}, acc1 = {};
  float rmax[8], rsum[8];
#pragma unroll
  for (int i = 0; i < 8; ++i) { rmax[i] = -1e30f; rsum[i] = 0.f; }

#if HAVE_ASYNC_LDS
  stage_chunk(kBase, C, &ldsK[0][0], lane);
  stage_chunk(vBase, HW, &ldsV[0][0], lane);
#endif

  for (int it = 0; it < HW / 32; ++it) {
    const int n0 = it * 32;
#if HAVE_ASYNC_LDS
    const int buf = it & 1;
    if (n0 + 32 < HW) {  // prefetch next chunk, then wait only for the current one
      stage_chunk(kBase + (size_t)(n0 + 32) * C, C, &ldsK[buf ^ 1][0], lane);
      stage_chunk(vBase + (n0 + 32), HW, &ldsV[buf ^ 1][0], lane);
      __builtin_amdgcn_s_wait_asynccnt(8);
    } else {
      __builtin_amdgcn_s_wait_asynccnt(0);
    }
    __syncthreads();
    const _Float16* kC = &ldsK[buf][0];
    const _Float16* vC = &ldsV[buf][0];
    const v16h kb0 = load_b32x16(kC + r * 32);
    const v16h kb1 = load_b32x16(kC + (16 + r) * 32);
#else
    const v16h kb0 = load_b32x16(kBase + (size_t)(n0 + r) * C);
    const v16h kb1 = load_b32x16(kBase + (size_t)(n0 + 16 + r) * C);
#endif
    v8f z = {};
    v8f s0 = WMMA_F16(qa, kb0, z);
    v8f s1 = WMMA_F16(qa, kb1, z);

#pragma unroll
    for (int i = 0; i < 8; ++i) {
      float a = s0[i] * scale, bb = s1[i] * scale;
      float t = fmaxf(a, bb);  // row = i + 8*hf lives on the 16 lanes of this half-wave
#pragma unroll
      for (int m = 1; m < 16; m <<= 1) t = fmaxf(t, __shfl_xor(t, m, 32));
      const float nm = fmaxf(rmax[i], t);
      const float corr = __expf(rmax[i] - nm);
      rmax[i] = nm;
      const float p0 = __expf(a - nm);
      const float p1 = __expf(bb - nm);
      float ps = p0 + p1;
#pragma unroll
      for (int m = 1; m < 16; m <<= 1) ps += __shfl_xor(ps, m, 32);
      rsum[i] = rsum[i] * corr + ps;
      acc0[i] *= corr;
      acc1[i] *= corr;
      ldsP[(i + 8 * hf) * 32 + r]      = (_Float16)p0;  // D layout -> LDS
      ldsP[(i + 8 * hf) * 32 + 16 + r] = (_Float16)p1;
    }
    __syncthreads();
    const v16h pa = load_a16x32(ldsP + r * 32);  // re-read P in A layout
#if HAVE_ASYNC_LDS
    const v16h vb0 = load_b32x16(vC + r * 32);
    const v16h vb1 = load_b32x16(vC + (16 + r) * 32);
#else
    const v16h vb0 = load_b32x16(vBase + (size_t)r * HW + n0);
    const v16h vb1 = load_b32x16(vBase + (size_t)(16 + r) * HW + n0);
#endif
    acc0 = WMMA_F16(pa, vb0, acc0);
    acc1 = WMMA_F16(pa, vb1, acc1);
    __syncthreads();
  }

#pragma unroll
  for (int i = 0; i < 8; ++i) {
    const int m = m0 + i + 8 * hf;
    const float inv = 1.0f / rsum[i];
    const int ch0 = h * HC + r, ch1 = h * HC + 16 + r;
    const float o0 = acc0[i] * inv + lepe_val(qT, rpe_w, rpe_b, b, ch0, m);
    const float o1 = acc1[i] * inv + lepe_val(qT, rpe_w, rpe_b, b, ch1, m);
    outT16[((size_t)b * HW + m) * C + ch0] = (_Float16)o0;
    outT16[((size_t)b * HW + m) * C + ch1] = (_Float16)o1;
  }
}

// ---------------------------------------------------------------------------
// Host-side orchestration
// ---------------------------------------------------------------------------
static inline size_t al256(size_t x) { return (x + 255) & ~(size_t)255; }

extern "C" void kernel_launch(void* const* d_in, const int* in_sizes, int n_in,
                              void* d_out, int out_size, void* d_ws, size_t ws_size,
                              hipStream_t stream) {
  (void)in_sizes; (void)n_in; (void)out_size; (void)ws_size;
  const float* x      = (const float*)d_in[0];
  const float* Wq     = (const float*)d_in[1];
  const float* bq     = (const float*)d_in[2];
  const float* Wk     = (const float*)d_in[3];
  const float* bk     = (const float*)d_in[4];
  const float* Wv     = (const float*)d_in[5];
  const float* bv     = (const float*)d_in[6];
  const float* Wo     = (const float*)d_in[7];
  const float* bo     = (const float*)d_in[8];
  const float* dw_w   = (const float*)d_in[9];
  const float* dw_b   = (const float*)d_in[10];
  const float* ln_g   = (const float*)d_in[11];
  const float* ln_b   = (const float*)d_in[12];
  const float* pw_w   = (const float*)d_in[13];
  const float* rpe_w  = (const float*)d_in[14];
  const float* rpe_b  = (const float*)d_in[15];
  float* y = (float*)d_out;

  char* ws = (char*)d_ws;
  size_t off = 0;
  _Float16* xT16  = (_Float16*)(ws + off); off = al256(off + (size_t)BATCH * HW * C * 2);
  _Float16* Wq16  = (_Float16*)(ws + off); off = al256(off + (size_t)C * C * 2);
  _Float16* Wk16  = (_Float16*)(ws + off); off = al256(off + (size_t)C * C * 2);
  _Float16* Wv16  = (_Float16*)(ws + off); off = al256(off + (size_t)C * C * 2);
  _Float16* Wo16  = (_Float16*)(ws + off); off = al256(off + (size_t)C * C * 2);
  float*    qTf   = (float*)   (ws + off); off = al256(off + (size_t)BATCH * HW * C * 4);
  _Float16* qT16  = (_Float16*)(ws + off); off = al256(off + (size_t)BATCH * HW * C * 2);
  float*    pos   = (float*)   (ws + off); off = al256(off + (size_t)BATCH * NG * HW * 2 * 4);
  _Float16* xsT16 = (_Float16*)(ws + off); off = al256(off + (size_t)BATCH * HW * C * 2);
  _Float16* kT16  = (_Float16*)(ws + off); off = al256(off + (size_t)BATCH * HW * C * 2);
  _Float16* vcm16 = (_Float16*)(ws + off); off = al256(off + (size_t)BATCH * HW * C * 2);
  _Float16* oT16  = (_Float16*)(ws + off); off = al256(off + (size_t)BATCH * HW * C * 2);

  const int nTot = BATCH * C * HW;                // 2M
  const int nW   = C * C;                         // 64K
  const int nPix = BATCH * NG * HW;               // 32K

  cvt_xT_kernel<<<(nTot + 255) / 256, 256, 0, stream>>>(x, xT16);
  cvt_f16_kernel<<<(nW + 255) / 256, 256, 0, stream>>>(Wq, Wq16, nW);
  cvt_f16_kernel<<<(nW + 255) / 256, 256, 0, stream>>>(Wk, Wk16, nW);
  cvt_f16_kernel<<<(nW + 255) / 256, 256, 0, stream>>>(Wv, Wv16, nW);
  cvt_f16_kernel<<<(nW + 255) / 256, 256, 0, stream>>>(Wo, Wo16, nW);

  dim3 gemmGrid(HW / 16, C / 64, BATCH);
  qproj_kernel<<<gemmGrid, 32, 0, stream>>>(xT16, Wq16, bq, qTf, qT16);

  offset_kernel<<<(nPix + 255) / 256, 256, 0, stream>>>(qTf, dw_w, dw_b, ln_g, ln_b, pw_w, pos);
  sample_kernel<<<(nPix + 255) / 256, 256, 0, stream>>>(x, pos, xsT16);

  kproj_kernel<<<gemmGrid, 32, 0, stream>>>(xsT16, Wk16, bk, kT16);
  vproj_kernel<<<gemmGrid, 32, 0, stream>>>(xsT16, Wv16, bv, vcm16);

  dim3 attnGrid(HW / 16, NH, BATCH);
  attn_kernel<<<attnGrid, 32, 0, stream>>>(qT16, kT16, vcm16, qTf, rpe_w, rpe_b, oT16);

  oproj_kernel<<<gemmGrid, 32, 0, stream>>>(oT16, Wo16, bo, y);
}
